// SelfAttention_2877628088475
// MI455X (gfx1250) — matrix operand order repaired and checked by reading, not compiled, not run on hardware
//
#include <hip/hip_runtime.h>
#include <hip/hip_bf16.h>

// Fused causal flash-attention fwd, (B,S,H,D) = (4,2048,16,64), fp32 I/O.
// Workgroup = 8 waves = 128 Q rows of one (b,h). KV walked in 64-row tiles:
//   - tile t+1 streamed to LDS with global_load_async_to_lds_b128 (ASYNCcnt)
//   - cooperative f32->f16 transform (K row-major, V transposed)
//   - per wave: QK^T + online softmax (DPP reductions) + P.V, 16 WMMA/tile

typedef __attribute__((ext_vector_type(16))) _Float16 v16h;
typedef __attribute__((ext_vector_type(8)))  _Float16 v8h;
typedef __attribute__((ext_vector_type(8)))  float    v8f;
typedef __attribute__((ext_vector_type(4)))  float    v4f;

#define WAVES 8

constexpr int S = 2048, H = 16, D = 64;
constexpr int ROWF = 3 * H * D;            // 3072 floats between consecutive s
constexpr int ROWB = ROWF * 4;             // 12288 bytes
constexpr int KT   = 64;                   // kv rows per tile

__device__ __forceinline__ unsigned ldsOff(const void* p) {
  return (unsigned)(uintptr_t)p;           // low 32 bits = LDS byte offset
}

__device__ __forceinline__ v16h ldsFrag(const _Float16* p) {
  const v8h a = *(const v8h*)p;
  const v8h b = *(const v8h*)(p + 16);
  v16h r;
#pragma unroll
  for (int i = 0; i < 8; ++i) { r[i] = a[i]; r[8 + i] = b[i]; }
  return r;
}

// 16-lane xor-reduction trees via DPP (no LDS crossbar):
// quad_perm xor1 (0xB1), quad_perm xor2 (0x4E), row_half_mirror (0x141 == xor4
// once quads are uniform), row_mirror (0x140 == xor8 once octets are uniform).
template <int CTRL>
__device__ __forceinline__ float dppMaxStep(float x) {
  const int m = __builtin_amdgcn_update_dpp(0, __float_as_int(x), CTRL, 0xf, 0xf, true);
  return fmaxf(x, __int_as_float(m));
}
template <int CTRL>
__device__ __forceinline__ float dppAddStep(float x) {
  const int m = __builtin_amdgcn_update_dpp(0, __float_as_int(x), CTRL, 0xf, 0xf, true);
  return x + __int_as_float(m);
}
__device__ __forceinline__ float rowMax16(float x) {
  x = dppMaxStep<0xB1>(x); x = dppMaxStep<0x4E>(x);
  x = dppMaxStep<0x141>(x); x = dppMaxStep<0x140>(x);
  return x;
}
__device__ __forceinline__ float rowSum16(float x) {
  x = dppAddStep<0xB1>(x); x = dppAddStep<0x4E>(x);
  x = dppAddStep<0x141>(x); x = dppAddStep<0x140>(x);
  return x;
}

__global__ __launch_bounds__(256) void
fa_fwd_causal_kernel(const float* __restrict__ qkv, float* __restrict__ out) {
  const int tid  = threadIdx.x;
  const int lane = tid & 31;
  const int wave = tid >> 5;
  const int nn   = lane & 15;
  const int g    = lane >> 4;

  const int b = blockIdx.z;
  const int h = blockIdx.y;
  const int qblk  = blockIdx.x * 128;
  const int qbase = qblk + wave * 16;

  const float* Qb = qkv + (size_t)b * S * ROWF + h * D;
  const float* Kb = Qb + H * D;
  const float* Vb = Qb + 2 * H * D;

  __shared__ __align__(16) float    rawK[2][KT * 64];   // async targets (f32)
  __shared__ __align__(16) float    rawV[2][KT * 64];
  __shared__ __align__(16) _Float16 k16[KT * 64];       // [kv][d]
  __shared__ __align__(16) _Float16 vt16[64 * KT];      // [d][kv] transposed
  __shared__ __align__(16) _Float16 sP[WAVES][16 * KT]; // per-wave P scratch
  _Float16* pw = &sP[wave][0];

  // ---- Q fragments (A layout), 1/sqrt(D) folded in (exact: power of two)
  v16h qA[2];
  {
    const float* qrow = Qb + (size_t)(qbase + nn) * ROWF;
#pragma unroll
    for (int c = 0; c < 2; ++c) {
      const v4f f0 = *(const v4f*)(qrow + c * 32 + 8 * g);
      const v4f f1 = *(const v4f*)(qrow + c * 32 + 8 * g + 4);
      const v4f f2 = *(const v4f*)(qrow + c * 32 + 16 + 8 * g);
      const v4f f3 = *(const v4f*)(qrow + c * 32 + 16 + 8 * g + 4);
#pragma unroll
      for (int i = 0; i < 4; ++i) {
        qA[c][i]      = (_Float16)(f0[i] * 0.125f);
        qA[c][4 + i]  = (_Float16)(f1[i] * 0.125f);
        qA[c][8 + i]  = (_Float16)(f2[i] * 0.125f);
        qA[c][12 + i] = (_Float16)(f3[i] * 0.125f);
      }
    }
  }

  // ---- async stage one 64-row K/V tile (1024 x 16B chunks each, GVS mode)
  auto issueTile = [&](int kvbase, int buf) {
    const unsigned kLds = ldsOff(&rawK[buf][0]);
    const unsigned vLds = ldsOff(&rawV[buf][0]);
    const unsigned long long kBase =
        (unsigned long long)(uintptr_t)(Kb + (size_t)kvbase * ROWF);
    const unsigned long long vBase =
        (unsigned long long)(uintptr_t)(Vb + (size_t)kvbase * ROWF);
#pragma unroll
    for (int j = 0; j < 4; ++j) {
      const unsigned c = (unsigned)tid + (unsigned)j * 256u;   // 0..1023
      const unsigned gOff = (c >> 4) * (unsigned)ROWB + (c & 15u) * 16u;
      const unsigned lOff = c * 16u;
      asm volatile("global_load_async_to_lds_b128 %0, %1, %2"
                   :: "v"(kLds + lOff), "v"(gOff), "s"(kBase) : "memory");
      asm volatile("global_load_async_to_lds_b128 %0, %1, %2"
                   :: "v"(vLds + lOff), "v"(gOff), "s"(vBase) : "memory");
    }
  };

  v8f o[4] = {{0}, {0}, {0}, {0}};
  float m_i[8], l_i[8];
#pragma unroll
  for (int r = 0; r < 8; ++r) { m_i[r] = -3.0e38f; l_i[r] = 0.0f; }

  const int nTiles = (qblk >> 6) + 2;      // kv rows 0 .. qblk+127
  issueTile(0, 0);

  for (int t = 0; t < nTiles; ++t) {
    const int kvbase = t * KT;
    if (t + 1 < nTiles) {
      issueTile((t + 1) * KT, (t + 1) & 1);
      asm volatile("s_wait_asynccnt 8" ::: "memory");   // tile t landed
    } else {
      asm volatile("s_wait_asynccnt 0" ::: "memory");
    }
    __syncthreads();

    // ---- f32 -> f16 transform: K row-major, V transposed (16 elts/thread)
    {
      const int row   = tid >> 2;          // 0..63
      const int dbase = (tid & 3) * 16;    // 0..48
      const float* sk = &rawK[t & 1][row * 64 + dbase];
      v8h hk0, hk1;
#pragma unroll
      for (int j = 0; j < 2; ++j) {
        const v4f a = *(const v4f*)(sk + j * 8);
        const v4f c = *(const v4f*)(sk + j * 8 + 4);
        v8h& hh = j ? hk1 : hk0;
#pragma unroll
        for (int i = 0; i < 4; ++i) { hh[i] = (_Float16)a[i]; hh[4 + i] = (_Float16)c[i]; }
      }
      *(v8h*)&k16[row * 64 + dbase]     = hk0;
      *(v8h*)&k16[row * 64 + dbase + 8] = hk1;

      const float* sv = &rawV[t & 1][row * 64 + dbase];
#pragma unroll
      for (int j = 0; j < 4; ++j) {
        const v4f u = *(const v4f*)(sv + j * 4);
#pragma unroll
        for (int i = 0; i < 4; ++i)
          vt16[(dbase + j * 4 + i) * KT + row] = (_Float16)u[i];
      }
    }
    __syncthreads();

    if (kvbase <= qbase + 15) {            // uniform per wave (causal range)
      // ---- S = (Q/sqrtD) K^T : four 16-col fragments, K-dim 64 in 2 chunks
      v8f s[4] = {{0}, {0}, {0}, {0}};
#pragma unroll
      for (int f = 0; f < 4; ++f) {
        const v16h kb0 = ldsFrag(&k16[(f * 16 + nn) * 64 + 8 * g]);
        const v16h kb1 = ldsFrag(&k16[(f * 16 + nn) * 64 + 32 + 8 * g]);
        s[f] = __builtin_amdgcn_wmma_f32_16x16x32_f16(false, qA[0], false, kb0, (short)0, s[f], false, false);
        s[f] = __builtin_amdgcn_wmma_f32_16x16x32_f16(false, qA[1], false, kb1, (short)0, s[f], false, false);
      }

      // ---- causal mask only on diagonal-crossing tiles (uniform branch)
      if (kvbase + KT - 1 > qbase) {
#pragma unroll
        for (int f = 0; f < 4; ++f)
#pragma unroll
          for (int r = 0; r < 8; ++r) {
            const int q = qbase + r + 8 * g;
            if (kvbase + f * 16 + nn > q) s[f][r] = -3.0e38f;
          }
      }

      // ---- online softmax (row = 16 lanes of a half-wave; DPP reductions)
#pragma unroll
      for (int r = 0; r < 8; ++r) {
        float mx = fmaxf(fmaxf(s[0][r], s[1][r]), fmaxf(s[2][r], s[3][r]));
        mx = rowMax16(mx);
        const float mnew  = fmaxf(m_i[r], mx);
        const float alpha = __expf(m_i[r] - mnew);
        float p0 = __expf(s[0][r] - mnew);
        float p1 = __expf(s[1][r] - mnew);
        float p2 = __expf(s[2][r] - mnew);
        float p3 = __expf(s[3][r] - mnew);
        const float rs = rowSum16((p0 + p1) + (p2 + p3));
        l_i[r] = l_i[r] * alpha + rs;
        m_i[r] = mnew;
        s[0][r] = p0; s[1][r] = p1; s[2][r] = p2; s[3][r] = p3;
#pragma unroll
        for (int k = 0; k < 4; ++k) o[k][r] *= alpha;
      }

      // ---- transpose P (16x64) through per-wave LDS scratch
#pragma unroll
      for (int r = 0; r < 8; ++r) {
        const int m = r + 8 * g;
#pragma unroll
        for (int f = 0; f < 4; ++f)
          pw[m * KT + f * 16 + nn] = (_Float16)s[f][r];
      }
      asm volatile("" ::: "memory");       // same-wave DS ops are in-order
      v16h pA[2];
#pragma unroll
      for (int a = 0; a < 2; ++a) {
        const v8h x1 = *(const v8h*)(pw + nn * KT + a * 32 + 8 * g);
        const v8h x2 = *(const v8h*)(pw + nn * KT + a * 32 + 16 + 8 * g);
#pragma unroll
        for (int i = 0; i < 8; ++i) { pA[a][i] = x1[i]; pA[a][8 + i] = x2[i]; }
      }

      // ---- O += P.V : 4 d-chunks x 2 kv-halves
#pragma unroll
      for (int n = 0; n < 4; ++n) {
        const v16h vb0 = ldsFrag(&vt16[(n * 16 + nn) * KT + 8 * g]);
        const v16h vb1 = ldsFrag(&vt16[(n * 16 + nn) * KT + 32 + 8 * g]);
        o[n] = __builtin_amdgcn_wmma_f32_16x16x32_f16(false, pA[0], false, vb0, (short)0, o[n], false, false);
        o[n] = __builtin_amdgcn_wmma_f32_16x16x32_f16(false, pA[1], false, vb1, (short)0, o[n], false, false);
      }
    }
  }

  // ---- normalize and store: out is (B,H,S,D)
  float* ob = out + (size_t)(b * H + h) * S * D;
#pragma unroll
  for (int r = 0; r < 8; ++r) {
    const float inv = 1.0f / l_i[r];
    const int row = qbase + r + 8 * g;
#pragma unroll
    for (int n = 0; n < 4; ++n)
      ob[(size_t)row * D + n * 16 + nn] = o[n][r] * inv;
  }
}

extern "C" void kernel_launch(void* const* d_in, const int* in_sizes, int n_in,
                              void* d_out, int out_size, void* d_ws, size_t ws_size,
                              hipStream_t stream) {
  (void)in_sizes; (void)n_in; (void)d_ws; (void)ws_size; (void)out_size;
  const float* qkv = (const float*)d_in[0];
  float* out = (float*)d_out;
  dim3 grid(S / 128, H, 4);        // (16, 16, 4): 128 Q rows per block
  dim3 block(32 * WAVES);          // 8 waves
  fa_fwd_causal_kernel<<<grid, block, 0, stream>>>(qkv, out);
}
